// TransformerDecoderLayer_32169305047725
// MI455X (gfx1250) — compile-verified, matrix operand order checked by reading, not run on hardware
//
#include <hip/hip_runtime.h>
#include <hip/hip_bf16.h>
#include <math.h>

typedef __bf16 bf16;
typedef __attribute__((ext_vector_type(16))) __bf16 v16bf;
typedef __attribute__((ext_vector_type(8)))  __bf16 v8bf;
typedef __attribute__((ext_vector_type(8)))  float  v8f;

#define DEV __device__ __forceinline__

static constexpr int Dm  = 512;     // d_model
static constexpr int Hh  = 8;       // heads
static constexpr int HDd = 64;      // head dim
static constexpr int HIDm= 2048;    // mlp hidden
static constexpr int Bb  = 8;
static constexpr int Nn  = 1024;
static constexpr int MT  = Bb * Nn; // 8192 rows
static constexpr float ATT_SCALE = 0.125f;   // 64^-0.5
static constexpr float LN_EPS = 1e-5f;

// ---------------------------------------------------------------------------
// Fused DPP16 butterfly reductions within a 16-lane row group.
// Single-instruction stages: v_max_num_f32_dpp / v_add_f32_dpp with src0
// routed through the DPP network (quad_perm xor1/xor2, half-mirror, mirror).
// ---------------------------------------------------------------------------
#define DPP_STAGE_F32(OPC, CTRL, dst, src)                                   \
  asm(OPC " %0, %1, %1 " CTRL " row_mask:0xf bank_mask:0xf bound_ctrl:1"     \
      : "=v"(dst) : "v"(src))

DEV float rowmax16(float v) {
  float t;
  DPP_STAGE_F32("v_max_num_f32_dpp", "quad_perm:[1,0,3,2]", t, v);
  DPP_STAGE_F32("v_max_num_f32_dpp", "quad_perm:[2,3,0,1]", v, t);
  DPP_STAGE_F32("v_max_num_f32_dpp", "row_half_mirror", t, v);
  DPP_STAGE_F32("v_max_num_f32_dpp", "row_mirror", v, t);
  return v;
}
DEV float rowsum16(float v) {
  float t;
  DPP_STAGE_F32("v_add_f32_dpp", "quad_perm:[1,0,3,2]", t, v);
  DPP_STAGE_F32("v_add_f32_dpp", "quad_perm:[2,3,0,1]", v, t);
  DPP_STAGE_F32("v_add_f32_dpp", "row_half_mirror", t, v);
  DPP_STAGE_F32("v_add_f32_dpp", "row_mirror", v, t);
  return v;
}

// ---------------------------------------------------------------------------
// Fragment helpers (CDNA5 16-bit WMMA layouts, wave32)
// A (16xK) / B^T (16 rows x K): lane&15 = row, halves split K into [ko..ko+7]
// and [ko+16..ko+23] with ko = (lane>>4)*8.  C/D f32: VGPR i -> M = i+8*(lane>=16),
// N = lane&15.
// ---------------------------------------------------------------------------
DEV v16bf load_frag(const bf16* base, int ld) {
  int lane = threadIdx.x & 31;
  int r  = lane & 15;
  int ko = (lane >> 4) << 3;
  const bf16* p = base + (size_t)r * ld + ko;
  v8bf lo = *(const v8bf*)(p);
  v8bf hi = *(const v8bf*)(p + 16);
  v16bf f;
#pragma unroll
  for (int i = 0; i < 8; ++i) { f[i] = lo[i]; f[8 + i] = hi[i]; }
  return f;
}

DEV v8f wmma_bf16(v16bf a, v16bf b, v8f c) {
  return __builtin_amdgcn_wmma_f32_16x16x32_bf16(false, a, false, b, (short)0, c,
                                                 false, false);
}

DEV float gelu_exact(float x) {
  return 0.5f * x * (1.0f + erff(x * 0.70710678118654752f));
}

// ---------------------------------------------------------------------------
// Elementwise prep kernels
// ---------------------------------------------------------------------------
__global__ void k_cvt_bf16(const float* __restrict__ in, bf16* __restrict__ obf,
                           float* __restrict__ fcopy, int n) {
  int i = blockIdx.x * 256 + threadIdx.x;
  if (i < n) {
    float v = in[i];
    obf[i] = (bf16)v;
    if (fcopy) fcopy[i] = v;
  }
}

// w[K,N] (f32, row-major) -> wt[N,K] (bf16)
__global__ void k_transpose_w(const float* __restrict__ w, bf16* __restrict__ wt,
                              int K, int Ncols) {
  int i = blockIdx.x * 256 + threadIdx.x;
  if (i < K * Ncols) {
    int n = i / K, k = i - n * K;
    wt[i] = (bf16)w[(size_t)k * Ncols + n];
  }
}

// ---------------------------------------------------------------------------
// GEMM: Out[M,N] = A[M,K] (bf16) * BT[N,K]^T (bf16), fp32 accumulate.
// Block 256 threads = 8 waves (4 in M x 2 in N); per-wave tile 32x64.
// MODE 0: store bf16, no bias. MODE 1: store f32 + bias. MODE 2: bf16 + bias + GELU.
// ---------------------------------------------------------------------------
template <int MODE>
__global__ void __launch_bounds__(256)
k_gemm(const bf16* __restrict__ A, const bf16* __restrict__ BT,
       const float* __restrict__ bias, void* __restrict__ Out,
       int M, int Ncols, int K) {
  int lane = threadIdx.x & 31, wave = threadIdx.x >> 5;
  int wm = wave >> 1, wn = wave & 1;
  int m0 = blockIdx.y * 128 + wm * 32;
  int n0 = blockIdx.x * 128 + wn * 64;

  v8f acc[2][4];
#pragma unroll
  for (int i = 0; i < 2; ++i)
#pragma unroll
    for (int j = 0; j < 4; ++j) acc[i][j] = {};

  const bf16* a0p = A + (size_t)m0 * K;
  const bf16* a1p = A + (size_t)(m0 + 16) * K;
  const bf16* bp0 = BT + (size_t)(n0)      * K;
  const bf16* bp1 = BT + (size_t)(n0 + 16) * K;
  const bf16* bp2 = BT + (size_t)(n0 + 32) * K;
  const bf16* bp3 = BT + (size_t)(n0 + 48) * K;

  for (int k0 = 0; k0 < K; k0 += 32) {
    v16bf a0 = load_frag(a0p + k0, K);
    v16bf a1 = load_frag(a1p + k0, K);
    v16bf b0 = load_frag(bp0 + k0, K);
    v16bf b1 = load_frag(bp1 + k0, K);
    v16bf b2 = load_frag(bp2 + k0, K);
    v16bf b3 = load_frag(bp3 + k0, K);
    acc[0][0] = wmma_bf16(a0, b0, acc[0][0]);
    acc[0][1] = wmma_bf16(a0, b1, acc[0][1]);
    acc[0][2] = wmma_bf16(a0, b2, acc[0][2]);
    acc[0][3] = wmma_bf16(a0, b3, acc[0][3]);
    acc[1][0] = wmma_bf16(a1, b0, acc[1][0]);
    acc[1][1] = wmma_bf16(a1, b1, acc[1][1]);
    acc[1][2] = wmma_bf16(a1, b2, acc[1][2]);
    acc[1][3] = wmma_bf16(a1, b3, acc[1][3]);
  }

  int hi = lane >> 4, nl = lane & 15;
#pragma unroll
  for (int i = 0; i < 2; ++i) {
#pragma unroll
    for (int j = 0; j < 4; ++j) {
#pragma unroll
      for (int e = 0; e < 8; ++e) {
        int gm = m0 + 16 * i + e + 8 * hi;
        int gn = n0 + 16 * j + nl;
        float v = acc[i][j][e];
        if (MODE >= 1) v += bias[gn];
        if (MODE == 2) v = gelu_exact(v);
        size_t idx = (size_t)gm * Ncols + gn;
        if (MODE == 1) ((float*)Out)[idx] = v;
        else           ((bf16*)Out)[idx]  = (bf16)v;
      }
    }
  }
}

// ---------------------------------------------------------------------------
// Flash attention (head dim 64, bf16 inputs, fp32 softmax).
// Block = 256 threads = 8 waves; each wave owns 16 q-rows -> 128 q-rows/block.
// Grid = B*H*(Nq/128). KV processed in 32-row tiles staged in LDS.
// Row reductions use single-instruction DPP16 butterflies.
// ---------------------------------------------------------------------------
__global__ void __launch_bounds__(256)
k_attn(const bf16* __restrict__ Qb, int ldq, int cq,
       const bf16* __restrict__ Kb, int ldk, int ck,
       const bf16* __restrict__ Vb, int ldv, int cv,
       bf16* __restrict__ Ob, int ldo, int Nq, int Nk) {
  __shared__ bf16 Kt[32 * 64];      // K tile: [kv 32][hd 64]
  __shared__ bf16 Vt[64 * 40];      // V^T tile: [hd 64][kv 32] (+pad)
  __shared__ bf16 Pt[8 * 16 * 32];  // per-wave P: [16 q][32 kv]

  int tid = threadIdx.x, lane = tid & 31, wave = tid >> 5;
  int qtiles = Nq >> 7;
  int qt = blockIdx.x % qtiles;
  int bh = blockIdx.x / qtiles;
  int b = bh / Hh, h = bh % Hh;

  int qrow = b * Nq + qt * 128 + wave * 16;
  const bf16* Qp = Qb + (size_t)qrow * ldq + cq + h * HDd;
  v16bf qa0 = load_frag(Qp, ldq);        // hd 0..31
  v16bf qa1 = load_frag(Qp + 32, ldq);   // hd 32..63

  v8f of[4];
#pragma unroll
  for (int j = 0; j < 4; ++j) of[j] = {};
  float rmax[8], rsum[8];
#pragma unroll
  for (int i = 0; i < 8; ++i) { rmax[i] = -INFINITY; rsum[i] = 0.0f; }

  const bf16* Kp = Kb + (size_t)(b * Nk) * ldk + ck + h * HDd;
  const bf16* Vp = Vb + (size_t)(b * Nk) * ldv + cv + h * HDd;
  int lrow = tid >> 3;          // 0..31 kv row in tile
  int lch  = (tid & 7) * 8;     // hd chunk
  bf16* Pw = Pt + wave * (16 * 32);
  int hi = lane >> 4, nl = lane & 15;

  for (int kt = 0; kt < Nk; kt += 32) {
    __syncthreads();  // previous iteration done with Kt/Vt
    *(v8bf*)(Kt + lrow * 64 + lch) =
        *(const v8bf*)(Kp + (size_t)(kt + lrow) * ldk + lch);
    v8bf vv = *(const v8bf*)(Vp + (size_t)(kt + lrow) * ldv + lch);
#pragma unroll
    for (int i = 0; i < 8; ++i) Vt[(lch + i) * 40 + lrow] = vv[i];
    __syncthreads();

    // S = Q K^T  (16 q x 32 kv, two 16-wide C frags)
    v8f s0 = {}, s1 = {};
    s0 = wmma_bf16(qa0, load_frag(Kt + 0 * 64 + 0, 64), s0);
    s0 = wmma_bf16(qa1, load_frag(Kt + 0 * 64 + 32, 64), s0);
    s1 = wmma_bf16(qa0, load_frag(Kt + 16 * 64 + 0, 64), s1);
    s1 = wmma_bf16(qa1, load_frag(Kt + 16 * 64 + 32, 64), s1);

#pragma unroll
    for (int i = 0; i < 8; ++i) {
      float a = s0[i] * ATT_SCALE;
      float c = s1[i] * ATT_SCALE;
      float mx = rowmax16(fmaxf(a, c));
      float nm = fmaxf(rmax[i], mx);
      float alpha = __expf(rmax[i] - nm);
      rmax[i] = nm;
      float p0 = __expf(a - nm);
      float p1 = __expf(c - nm);
      float ss = rowsum16(p0 + p1);
      rsum[i] = rsum[i] * alpha + ss;
#pragma unroll
      for (int j = 0; j < 4; ++j) of[j][i] *= alpha;
      int m = i + 8 * hi;
      Pw[m * 32 + nl]      = (bf16)p0;
      Pw[m * 32 + 16 + nl] = (bf16)p1;
    }
    __syncthreads();  // P visible across lanes

    v16bf pf = load_frag(Pw, 32);
#pragma unroll
    for (int j = 0; j < 4; ++j)
      of[j] = wmma_bf16(pf, load_frag(Vt + (16 * j) * 40, 40), of[j]);
  }

  // O normalize + store (bf16, [row, h*64 + col])
  float rinv[8];
#pragma unroll
  for (int e = 0; e < 8; ++e) rinv[e] = 1.0f / rsum[e];
#pragma unroll
  for (int j = 0; j < 4; ++j) {
#pragma unroll
    for (int e = 0; e < 8; ++e) {
      int m = e + 8 * hi;
      float v = of[j][e] * rinv[e];
      Ob[(size_t)(qrow + m) * ldo + h * HDd + 16 * j + nl] = (bf16)v;
    }
  }
}

// ---------------------------------------------------------------------------
// Fused residual add + LayerNorm over D=512. One block (256 thr) per row.
// FINAL=0: res <- y, obf <- bf16(y).  FINAL=1: ofin <- y only.
// DPP 16-group reduce, then combine the 16 groups through LDS.
// ---------------------------------------------------------------------------
template <int FINAL>
__global__ void __launch_bounds__(256)
k_add_ln(float* __restrict__ res, const float* __restrict__ delta,
         const float* __restrict__ g, const float* __restrict__ bt,
         bf16* __restrict__ obf, float* __restrict__ ofin) {
  __shared__ float sm[32];
  int tid = threadIdx.x;
  size_t base = (size_t)blockIdx.x * Dm;
  float v0 = res[base + tid]       + delta[base + tid];
  float v1 = res[base + 256 + tid] + delta[base + 256 + tid];
  float s = rowsum16(v0 + v1);
  float q = rowsum16(v0 * v0 + v1 * v1);
  if ((tid & 15) == 0) {
    sm[tid >> 4]        = s;   // 16 groups
    sm[16 + (tid >> 4)] = q;
  }
  __syncthreads();
  if (tid == 0) {
    float ts = 0.f, tq = 0.f;
#pragma unroll
    for (int w = 0; w < 16; ++w) { ts += sm[w]; tq += sm[16 + w]; }
    sm[0] = ts; sm[16] = tq;
  }
  __syncthreads();
  float mean = sm[0] * (1.0f / Dm);
  float var  = sm[16] * (1.0f / Dm) - mean * mean;
  float rstd = rsqrtf(var + LN_EPS);
  float y0 = (v0 - mean) * rstd * g[tid]       + bt[tid];
  float y1 = (v1 - mean) * rstd * g[256 + tid] + bt[256 + tid];
  if (FINAL) {
    ofin[base + tid] = y0;
    ofin[base + 256 + tid] = y1;
  } else {
    res[base + tid] = y0;
    res[base + 256 + tid] = y1;
    obf[base + tid] = (bf16)y0;
    obf[base + 256 + tid] = (bf16)y1;
  }
}

// ---------------------------------------------------------------------------
extern "C" void kernel_launch(void* const* d_in, const int* in_sizes, int n_in,
                              void* d_out, int out_size, void* d_ws, size_t ws_size,
                              hipStream_t stream) {
  (void)in_sizes; (void)n_in; (void)out_size; (void)ws_size;
  const float* x      = (const float*)d_in[0];
  const float* y      = (const float*)d_in[1];
  const float* w_qkv  = (const float*)d_in[2];
  const float* w_p1   = (const float*)d_in[3];
  const float* b_p1   = (const float*)d_in[4];
  const float* w_q    = (const float*)d_in[5];
  const float* w_kv   = (const float*)d_in[6];
  const float* w_p2   = (const float*)d_in[7];
  const float* b_p2   = (const float*)d_in[8];
  const float* w_fc1  = (const float*)d_in[9];
  const float* b_fc1  = (const float*)d_in[10];
  const float* w_fc2  = (const float*)d_in[11];
  const float* b_fc2  = (const float*)d_in[12];
  const float* g1 = (const float*)d_in[13]; const float* bt1 = (const float*)d_in[14];
  const float* g2 = (const float*)d_in[15]; const float* bt2 = (const float*)d_in[16];
  const float* g3 = (const float*)d_in[17]; const float* bt3 = (const float*)d_in[18];

  char* ws = (char*)d_ws;
  size_t off = 0;
  auto alloc = [&](size_t bytes) -> void* {
    void* p = ws + off;
    off += (bytes + 255) & ~(size_t)255;
    return p;
  };

  bf16* wt_qkv = (bf16*)alloc((size_t)Dm * 3 * Dm * 2);
  bf16* wt_p1  = (bf16*)alloc((size_t)Dm * Dm * 2);
  bf16* wt_q   = (bf16*)alloc((size_t)Dm * Dm * 2);
  bf16* wt_kv  = (bf16*)alloc((size_t)Dm * 2 * Dm * 2);
  bf16* wt_p2  = (bf16*)alloc((size_t)Dm * Dm * 2);
  bf16* wt_fc1 = (bf16*)alloc((size_t)Dm * HIDm * 2);
  bf16* wt_fc2 = (bf16*)alloc((size_t)HIDm * Dm * 2);
  bf16* xbf    = (bf16*)alloc((size_t)MT * Dm * 2);
  bf16* ybf    = (bf16*)alloc((size_t)MT * Dm * 2);
  float* xres  = (float*)alloc((size_t)MT * Dm * 4);
  bf16* qkv    = (bf16*)alloc((size_t)MT * 3 * Dm * 2);
  bf16* att    = (bf16*)alloc((size_t)MT * Dm * 2);
  float* proj  = (float*)alloc((size_t)MT * Dm * 4);
  bf16* xlnbf  = (bf16*)alloc((size_t)MT * Dm * 2);
  bf16* q2b    = (bf16*)alloc((size_t)MT * Dm * 2);
  bf16* kvb    = (bf16*)alloc((size_t)MT * 2 * Dm * 2);
  bf16* hbf    = (bf16*)alloc((size_t)MT * HIDm * 2);

  auto cdiv = [](int a, int b) { return (a + b - 1) / b; };

  // Weight transposes (f32 [K,N] -> bf16 [N,K])
  k_transpose_w<<<cdiv(Dm * 3 * Dm, 256), 256, 0, stream>>>(w_qkv, wt_qkv, Dm, 3 * Dm);
  k_transpose_w<<<cdiv(Dm * Dm, 256),     256, 0, stream>>>(w_p1,  wt_p1,  Dm, Dm);
  k_transpose_w<<<cdiv(Dm * Dm, 256),     256, 0, stream>>>(w_q,   wt_q,   Dm, Dm);
  k_transpose_w<<<cdiv(Dm * 2 * Dm, 256), 256, 0, stream>>>(w_kv,  wt_kv,  Dm, 2 * Dm);
  k_transpose_w<<<cdiv(Dm * Dm, 256),     256, 0, stream>>>(w_p2,  wt_p2,  Dm, Dm);
  k_transpose_w<<<cdiv(Dm * HIDm, 256),   256, 0, stream>>>(w_fc1, wt_fc1, Dm, HIDm);
  k_transpose_w<<<cdiv(HIDm * Dm, 256),   256, 0, stream>>>(w_fc2, wt_fc2, HIDm, Dm);

  // Activation conversion (x also copied as the f32 running residual)
  k_cvt_bf16<<<cdiv(MT * Dm, 256), 256, 0, stream>>>(x, xbf, xres, MT * Dm);
  k_cvt_bf16<<<cdiv(MT * Dm, 256), 256, 0, stream>>>(y, ybf, nullptr, MT * Dm);

  // --- self attention ---
  k_gemm<0><<<dim3(3 * Dm / 128, MT / 128), 256, 0, stream>>>(
      xbf, wt_qkv, nullptr, qkv, MT, 3 * Dm, Dm);
  k_attn<<<Bb * Hh * (Nn / 128), 256, 0, stream>>>(
      qkv, 3 * Dm, 0,        // Q at cols [0,512)
      qkv, 3 * Dm, Dm,       // K at cols [512,1024)
      qkv, 3 * Dm, 2 * Dm,   // V at cols [1024,1536)
      att, Dm, Nn, Nn);
  k_gemm<1><<<dim3(Dm / 128, MT / 128), 256, 0, stream>>>(
      att, wt_p1, b_p1, proj, MT, Dm, Dm);
  k_add_ln<0><<<MT, 256, 0, stream>>>(xres, proj, g1, bt1, xlnbf, nullptr);

  // --- cross attention ---
  k_gemm<0><<<dim3(Dm / 128, MT / 128), 256, 0, stream>>>(
      xlnbf, wt_q, nullptr, q2b, MT, Dm, Dm);
  k_gemm<0><<<dim3(2 * Dm / 128, MT / 128), 256, 0, stream>>>(
      ybf, wt_kv, nullptr, kvb, MT, 2 * Dm, Dm);
  k_attn<<<Bb * Hh * (Nn / 128), 256, 0, stream>>>(
      q2b, Dm, 0,
      kvb, 2 * Dm, 0,
      kvb, 2 * Dm, Dm,
      att, Dm, Nn, Nn);
  k_gemm<1><<<dim3(Dm / 128, MT / 128), 256, 0, stream>>>(
      att, wt_p2, b_p2, proj, MT, Dm, Dm);
  k_add_ln<0><<<MT, 256, 0, stream>>>(xres, proj, g2, bt2, xlnbf, nullptr);

  // --- MLP ---
  k_gemm<2><<<dim3(HIDm / 128, MT / 128), 256, 0, stream>>>(
      xlnbf, wt_fc1, b_fc1, hbf, MT, HIDm, Dm);
  k_gemm<1><<<dim3(Dm / 128, MT / 128), 256, 0, stream>>>(
      hbf, wt_fc2, b_fc2, proj, MT, Dm, HIDm);
  k_add_ln<1><<<MT, 256, 0, stream>>>(xres, proj, g3, bt3, nullptr, (float*)d_out);
}
